// _EPScatterMoeForward_27693949125160
// MI455X (gfx1250) — matrix operand order repaired and checked by reading, not compiled, not run on hardware
//
#include <hip/hip_runtime.h>
#include <hip/hip_bf16.h>
#include <stdint.h>

// ---------------------------------------------------------------------------
// MoE forward for MI455X (gfx1250, wave32, WMMA).
// Grouped GEMMs in bf16 via v_wmma_f32_16x16x32_bf16, f32 accumulate.
// Double-buffered LDS staging; async global->LDS copies (ASYNCcnt path).
// ---------------------------------------------------------------------------

#define TOPK      8
#define NEXPERT   32
#define CAP       1024
#define HDIM      1024
#define FDIM      768
#define NTOK      2048            // B*S = 2*1024

typedef unsigned short ushort_t;
typedef __attribute__((ext_vector_type(16))) __bf16        v16bf;
typedef __attribute__((ext_vector_type(8)))  float         v8f;
typedef __attribute__((ext_vector_type(4)))  unsigned int  u32x4;
typedef __attribute__((ext_vector_type(4)))  float         f32x4;
typedef __attribute__((ext_vector_type(4)))  unsigned short us4;

// ---- gfx1250 async global->LDS copy (feature-detected) ---------------------
#if defined(__has_builtin)
#  if __has_builtin(__builtin_amdgcn_global_load_async_to_lds_b128) && \
      __has_builtin(__builtin_amdgcn_s_wait_asynccnt)
#    define USE_ASYNC_COPY 1
#  endif
#endif

#ifdef USE_ASYNC_COPY
// Builtin signature (from clang diagnostic): param0 = v4i in AS(1) [global],
// param1 = v4i in AS(3) [LDS], then imm offset, imm cpol.
typedef int v4i_vs __attribute__((vector_size(16)));
typedef __attribute__((address_space(1))) v4i_vs glob_v4i;
typedef __attribute__((address_space(3))) v4i_vs lds_v4i;
__device__ __forceinline__ void cp16_async(void* lds_dst, const void* gsrc) {
    __builtin_amdgcn_global_load_async_to_lds_b128(
        (glob_v4i*)gsrc, (lds_v4i*)lds_dst, 0, 0);
}
#endif

__device__ __forceinline__ unsigned short f2bf(float f) {
    unsigned int u = __float_as_uint(f);
    u += 0x7FFFu + ((u >> 16) & 1u);          // round-to-nearest-even
    return (unsigned short)(u >> 16);
}

__device__ __forceinline__ v16bf ld_frag_split(const ushort_t* lo, const ushort_t* hi) {
    union { u32x4 q[2]; v16bf v; } f;
    f.q[0] = *(const u32x4*)lo;               // 16B LDS load (b128)
    f.q[1] = *(const u32x4*)hi;               // 16B LDS load (b128)
    return f.v;
}
__device__ __forceinline__ v16bf ld_frag32(const ushort_t* p) {
    union { u32x4 q[2]; v16bf v; } f;
    f.q[0] = *(const u32x4*)p;
    f.q[1] = *(const u32x4*)(p + 8);
    return f.v;
}

// ---------------------------------------------------------------------------
// Kernel 1: zero output + per-expert counts.
// ---------------------------------------------------------------------------
__global__ __launch_bounds__(256) void k_init(float* __restrict__ out,
                                              int* __restrict__ counts, int n4) {
    int i = blockIdx.x * 256 + threadIdx.x;
    if (blockIdx.x == 0 && threadIdx.x < NEXPERT) counts[threadIdx.x] = 0;
    if (i < n4) {
        f32x4 z = 0.f;
        ((f32x4*)out)[i] = z;
    }
}

// ---------------------------------------------------------------------------
// Kernel 2: router. One wave per token; lane == expert (E=32 == wave32).
// ---------------------------------------------------------------------------
__global__ __launch_bounds__(256)
void k_router(const float* __restrict__ x, const float* __restrict__ gw,
              int* __restrict__ counts,
              int* __restrict__ slot_e, int* __restrict__ slot_pos,
              int* __restrict__ slot_token, float* __restrict__ slot_w) {
    const int lane = threadIdx.x & 31;
    const int wv   = threadIdx.x >> 5;
    const int tok  = blockIdx.x * 8 + wv;
    if (tok >= NTOK) return;

    const float* xrow = x  + (size_t)tok  * HDIM;
    const float* grow = gw + (size_t)lane * HDIM;

    float acc = 0.f;
    for (int h0 = 0; h0 < HDIM; h0 += 32) {
        float xv = xrow[h0 + lane];
#pragma unroll
        for (int j = 0; j < 32; ++j)
            acc = fmaf(__shfl(xv, j, 32), grow[h0 + j], acc);
    }

    // fp32 softmax across the wave
    float m = acc;
#pragma unroll
    for (int off = 16; off; off >>= 1) m = fmaxf(m, __shfl_xor(m, off, 32));
    float p = __expf(acc - m);
    float s = p;
#pragma unroll
    for (int off = 16; off; off >>= 1) s += __shfl_xor(s, off, 32);
    float prob = p / s;

    // iterative top-8 (wave-uniform results)
    float v = prob;
    float topv[TOPK]; int tope[TOPK];
#pragma unroll
    for (int k = 0; k < TOPK; ++k) {
        float mv = v; int mi = lane;
#pragma unroll
        for (int off = 16; off; off >>= 1) {
            float om = __shfl_xor(mv, off, 32);
            int   oi = __shfl_xor(mi, off, 32);
            if (om > mv || (om == mv && oi < mi)) { mv = om; mi = oi; }
        }
        topv[k] = mv; tope[k] = mi;
        if (lane == mi) v = -1.f;
    }
    float wsum = 0.f;
#pragma unroll
    for (int k = 0; k < TOPK; ++k) wsum += topv[k];

    if (lane == 0) {
#pragma unroll
        for (int k = 0; k < TOPK; ++k) {
            const int   ex = tope[k];
            const float wk = topv[k] / wsum;
            const int   ps = atomicAdd(&counts[ex], 1);
            const int   sl = tok * TOPK + k;
            slot_e[sl]   = ex;
            slot_pos[sl] = ps;
            if (ps < CAP) {
                slot_token[ex * CAP + ps] = tok;
                slot_w[ex * CAP + ps]     = wk;
            }
        }
    }
}

// ---------------------------------------------------------------------------
// Kernel 3: f32 -> bf16 conversion (weights), vectorized.
// ---------------------------------------------------------------------------
__global__ __launch_bounds__(256)
void k_f32_to_bf16(const float* __restrict__ src, ushort_t* __restrict__ dst, int n4) {
    for (int i = blockIdx.x * blockDim.x + threadIdx.x; i < n4;
         i += gridDim.x * blockDim.x) {
        f32x4 v = ((const f32x4*)src)[i];
        us4 o;
        o.x = f2bf(v.x); o.y = f2bf(v.y); o.z = f2bf(v.z); o.w = f2bf(v.w);
        ((us4*)dst)[i] = o;
    }
}

// ---------------------------------------------------------------------------
// Kernel 4: dispatch token rows into per-expert bf16 buffers [E, CAP, H].
// ---------------------------------------------------------------------------
__global__ __launch_bounds__(256)
void k_dispatch(const float* __restrict__ x,
                const int* __restrict__ slot_e, const int* __restrict__ slot_pos,
                ushort_t* __restrict__ xbuf) {
    const int sl  = blockIdx.x;          // 0 .. NTOK*TOPK-1
    const int tok = sl >> 3;
    const int e   = slot_e[sl];
    const int ps  = slot_pos[sl];
    if (ps >= CAP) return;
    const f32x4* src = (const f32x4*)(x + (size_t)tok * HDIM);
    us4* dst = (us4*)(xbuf + ((size_t)e * CAP + ps) * HDIM);
    const int i = threadIdx.x;           // H/4 == 256 == blockDim
    f32x4 v = src[i];
    us4 o;
    o.x = f2bf(v.x); o.y = f2bf(v.y); o.z = f2bf(v.z); o.w = f2bf(v.w);
    dst[i] = o;
}

// ---------------------------------------------------------------------------
// GEMM tiling (kernels 5/6): block = 256 threads (8 waves),
// tile BM=128 x BN=64, K-step 32, double-buffered LDS.
// Wave w owns rows [16w,16w+16); 4 accumulators of 16x16 per B-half.
// ---------------------------------------------------------------------------
#define BM 128
#define BN 64
#define BK 32
#define LSTR 56   // LDS row stride (halfwords): 112B, 16B-aligned, conflict-spread

// Kernel 5: h = silu(xbuf @ Wg^T) * (xbuf @ Wu^T), bf16 out.
__global__ __launch_bounds__(256)
void k_gemm1_silu(const ushort_t* __restrict__ xbuf,
                  const ushort_t* __restrict__ wgu,
                  ushort_t* __restrict__ hbuf) {
    const int e   = blockIdx.z;
    const int m0  = blockIdx.y * BM;
    const int n0  = blockIdx.x * BN;
    const int tid = threadIdx.x;
    const int lane = tid & 31;
    const int wv   = tid >> 5;

    __shared__ ushort_t As[2][BM * LSTR];
    __shared__ ushort_t Bg[2][BN * LSTR];
    __shared__ ushort_t Bu[2][BN * LSTR];

    const size_t abase = ((size_t)e * CAP + m0) * HDIM;
    const size_t gbase = ((size_t)e * (2 * FDIM) + n0) * HDIM;
    const size_t ubase = ((size_t)e * (2 * FDIM) + FDIM + n0) * HDIM;

    const int brow = tid >> 2, bcg = (tid & 3) * 8;

    auto load_tiles = [&](int buf, int k0) {
#ifdef USE_ASYNC_COPY
#pragma unroll
        for (int i = 0; i < 2; ++i) {
            const int c = tid + 256 * i;
            const int r = c >> 2, cgp = (c & 3) * 8;
            cp16_async(&As[buf][r * LSTR + cgp],
                       &xbuf[abase + (size_t)r * HDIM + k0 + cgp]);
        }
        cp16_async(&Bg[buf][brow * LSTR + bcg],
                   &wgu[gbase + (size_t)brow * HDIM + k0 + bcg]);
        cp16_async(&Bu[buf][brow * LSTR + bcg],
                   &wgu[ubase + (size_t)brow * HDIM + k0 + bcg]);
#else
#pragma unroll
        for (int i = 0; i < 2; ++i) {
            const int c = tid + 256 * i;
            const int r = c >> 2, cgp = (c & 3) * 8;
            *(u32x4*)&As[buf][r * LSTR + cgp] =
                *(const u32x4*)&xbuf[abase + (size_t)r * HDIM + k0 + cgp];
        }
        *(u32x4*)&Bg[buf][brow * LSTR + bcg] =
            *(const u32x4*)&wgu[gbase + (size_t)brow * HDIM + k0 + bcg];
        *(u32x4*)&Bu[buf][brow * LSTR + bcg] =
            *(const u32x4*)&wgu[ubase + (size_t)brow * HDIM + k0 + bcg];
        __builtin_prefetch(&wgu[gbase + (size_t)brow * HDIM + k0 + BK + bcg], 0, 1);
        __builtin_prefetch(&wgu[ubase + (size_t)brow * HDIM + k0 + BK + bcg], 0, 1);
#endif
    };

    v8f cg[4], cu[4];
#pragma unroll
    for (int j = 0; j < 4; ++j) { cg[j] = 0.f; cu[j] = 0.f; }

    const int aM    = lane & 15;
    const int akh   = (lane >> 4) * 16;   // A K-half start (elements)
    const int bkh   = (lane >> 4) * 16;   // B K-half start (elements)
    const int laneN = lane & 15;

    load_tiles(0, 0);
    const int nk = HDIM / BK;
    for (int ki = 0; ki < nk; ++ki) {
        const int buf = ki & 1;
#ifdef USE_ASYNC_COPY
        __builtin_amdgcn_s_wait_asynccnt(0);
#endif
        __syncthreads();                       // tile ready; prev reads done
        if (ki + 1 < nk) load_tiles(buf ^ 1, (ki + 1) * BK);

        // fragments: A once, all 8 B fragments, then 8 back-to-back WMMAs
        const ushort_t* ar = &As[buf][(wv * 16 + aM) * LSTR + akh];
        v16bf a = ld_frag32(ar);
        v16bf fb_g[4], fb_u[4];
#pragma unroll
        for (int j = 0; j < 4; ++j) {
            const int bn = j * 16 + laneN;
            fb_g[j] = ld_frag32(&Bg[buf][bn * LSTR + bkh]);
            fb_u[j] = ld_frag32(&Bu[buf][bn * LSTR + bkh]);
        }
#pragma unroll
        for (int j = 0; j < 4; ++j)
            cg[j] = __builtin_amdgcn_wmma_f32_16x16x32_bf16(
                        false, a, false, fb_g[j], (short)0, cg[j], false, false);
#pragma unroll
        for (int j = 0; j < 4; ++j)
            cu[j] = __builtin_amdgcn_wmma_f32_16x16x32_bf16(
                        false, a, false, fb_u[j], (short)0, cu[j], false, false);
    }

    // epilogue: silu(gate)*up -> bf16 hbuf
    const int laneM8 = (lane >> 4) * 8;
#pragma unroll
    for (int j = 0; j < 4; ++j) {
#pragma unroll
        for (int i = 0; i < 8; ++i) {
            const float g  = cg[j][i];
            const float u  = cu[j][i];
            const float hv = (g / (1.f + __expf(-g))) * u;
            const int row = m0 + wv * 16 + i + laneM8;
            const int col = n0 + j * 16 + laneN;
            hbuf[((size_t)e * CAP + row) * FDIM + col] = f2bf(hv);
        }
    }
}

// Kernel 6: eo = h @ Wd^T; fused weighted scatter-add into out[token].
__global__ __launch_bounds__(256)
void k_gemm2_scatter(const ushort_t* __restrict__ hbuf,
                     const ushort_t* __restrict__ wdn,
                     const int* __restrict__ counts,
                     const int* __restrict__ slot_token,
                     const float* __restrict__ slot_w,
                     float* __restrict__ out) {
    const int e   = blockIdx.z;
    const int m0  = blockIdx.y * BM;
    const int n0  = blockIdx.x * BN;
    const int tid = threadIdx.x;
    const int lane = tid & 31;
    const int wv   = tid >> 5;

    __shared__ ushort_t As[2][BM * LSTR];
    __shared__ ushort_t Bs[2][BN * LSTR];
    __shared__ int   s_tok[BM];
    __shared__ float s_w[BM];

    if (tid < BM) {
        s_tok[tid] = slot_token[e * CAP + m0 + tid];
        s_w[tid]   = slot_w[e * CAP + m0 + tid];
    }
    const int cnt = counts[e];

    const size_t abase = ((size_t)e * CAP + m0) * FDIM;
    const size_t bbase = ((size_t)e * HDIM + n0) * FDIM;

    const int brow = tid >> 2, bcg = (tid & 3) * 8;

    auto load_tiles = [&](int buf, int k0) {
#ifdef USE_ASYNC_COPY
#pragma unroll
        for (int i = 0; i < 2; ++i) {
            const int c = tid + 256 * i;
            const int r = c >> 2, cgp = (c & 3) * 8;
            cp16_async(&As[buf][r * LSTR + cgp],
                       &hbuf[abase + (size_t)r * FDIM + k0 + cgp]);
        }
        cp16_async(&Bs[buf][brow * LSTR + bcg],
                   &wdn[bbase + (size_t)brow * FDIM + k0 + bcg]);
#else
#pragma unroll
        for (int i = 0; i < 2; ++i) {
            const int c = tid + 256 * i;
            const int r = c >> 2, cgp = (c & 3) * 8;
            *(u32x4*)&As[buf][r * LSTR + cgp] =
                *(const u32x4*)&hbuf[abase + (size_t)r * FDIM + k0 + cgp];
        }
        *(u32x4*)&Bs[buf][brow * LSTR + bcg] =
            *(const u32x4*)&wdn[bbase + (size_t)brow * FDIM + k0 + bcg];
        __builtin_prefetch(&wdn[bbase + (size_t)brow * FDIM + k0 + BK + bcg], 0, 1);
#endif
    };

    v8f cc[4];
#pragma unroll
    for (int j = 0; j < 4; ++j) cc[j] = 0.f;

    const int aM    = lane & 15;
    const int akh   = (lane >> 4) * 16;
    const int bkh   = (lane >> 4) * 16;
    const int laneN = lane & 15;

    load_tiles(0, 0);
    const int nk = FDIM / BK;
    for (int ki = 0; ki < nk; ++ki) {
        const int buf = ki & 1;
#ifdef USE_ASYNC_COPY
        __builtin_amdgcn_s_wait_asynccnt(0);
#endif
        __syncthreads();
        if (ki + 1 < nk) load_tiles(buf ^ 1, (ki + 1) * BK);

        const ushort_t* ar = &As[buf][(wv * 16 + aM) * LSTR + akh];
        v16bf a = ld_frag32(ar);
        v16bf fb[4];
#pragma unroll
        for (int j = 0; j < 4; ++j)
            fb[j] = ld_frag32(&Bs[buf][(j * 16 + laneN) * LSTR + bkh]);
#pragma unroll
        for (int j = 0; j < 4; ++j)
            cc[j] = __builtin_amdgcn_wmma_f32_16x16x32_bf16(
                        false, a, false, fb[j], (short)0, cc[j], false, false);
    }

    // epilogue: out[tok, col] += w * val  (guarded against unused capacity rows)
    const int laneM8 = (lane >> 4) * 8;
#pragma unroll
    for (int j = 0; j < 4; ++j) {
#pragma unroll
        for (int i = 0; i < 8; ++i) {
            const int rloc = wv * 16 + i + laneM8;
            const int row  = m0 + rloc;
            if (row < cnt) {
                const int   tk = s_tok[rloc];
                const float w  = s_w[rloc];
                atomicAdd(&out[(size_t)tk * HDIM + n0 + j * 16 + laneN], w * cc[j][i]);
            }
        }
    }
}

// ---------------------------------------------------------------------------
// Host launcher
// ---------------------------------------------------------------------------
extern "C" void kernel_launch(void* const* d_in, const int* in_sizes, int n_in,
                              void* d_out, int out_size, void* d_ws, size_t ws_size,
                              hipStream_t stream) {
    const float* x   = (const float*)d_in[0];   // [2,1024,1024]
    const float* gw  = (const float*)d_in[1];   // [32,1024]
    const float* wgu = (const float*)d_in[2];   // [32,1536,1024]
    const float* wdn = (const float*)d_in[3];   // [32,1024,768]
    float* out = (float*)d_out;                 // [2,1024,1024]

    // workspace layout (256B-aligned slabs)
    char* base = (char*)d_ws;
    size_t off = 0;
    auto take = [&](size_t bytes) -> char* {
        char* p = base + off;
        off = (off + bytes + 255) & ~(size_t)255;
        return p;
    };
    int*      counts     = (int*)   take(NEXPERT * sizeof(int));
    int*      slot_e     = (int*)   take((size_t)NTOK * TOPK * sizeof(int));
    int*      slot_pos   = (int*)   take((size_t)NTOK * TOPK * sizeof(int));
    int*      slot_token = (int*)   take((size_t)NEXPERT * CAP * sizeof(int));
    float*    slot_w     = (float*) take((size_t)NEXPERT * CAP * sizeof(float));
    ushort_t* wgu_bf     = (ushort_t*)take((size_t)NEXPERT * 2 * FDIM * HDIM * 2);
    ushort_t* wdn_bf     = (ushort_t*)take((size_t)NEXPERT * HDIM * FDIM * 2);
    ushort_t* xbuf       = (ushort_t*)take((size_t)NEXPERT * CAP * HDIM * 2);
    ushort_t* hbuf       = (ushort_t*)take((size_t)NEXPERT * CAP * FDIM * 2);
    (void)ws_size; (void)in_sizes; (void)n_in; (void)out_size;

    // 1) zero out + counts
    k_init<<<(NTOK * HDIM / 4 + 255) / 256, 256, 0, stream>>>(out, counts,
                                                              NTOK * HDIM / 4);
    // 2) router
    k_router<<<NTOK / 8, 256, 0, stream>>>(x, gw, counts, slot_e, slot_pos,
                                           slot_token, slot_w);
    // 3/4) weight conversion to bf16 (144MB total -> L2-resident for GEMMs)
    k_f32_to_bf16<<<4096, 256, 0, stream>>>(wgu, wgu_bf,
                                            NEXPERT * 2 * FDIM * HDIM / 4);
    k_f32_to_bf16<<<4096, 256, 0, stream>>>(wdn, wdn_bf,
                                            NEXPERT * HDIM * FDIM / 4);
    // 5) dispatch
    k_dispatch<<<NTOK * TOPK, 256, 0, stream>>>(x, slot_e, slot_pos, xbuf);
    // 6) grouped GEMM1 + SiLU  (grid: 768/64 x 1024/128 x E)
    k_gemm1_silu<<<dim3(FDIM / BN, CAP / BM, NEXPERT), 256, 0, stream>>>(
        xbuf, wgu_bf, hbuf);
    // 7) grouped GEMM2 + weighted scatter-add (grid: 1024/64 x 1024/128 x E)
    k_gemm2_scatter<<<dim3(HDIM / BN, CAP / BM, NEXPERT), 256, 0, stream>>>(
        hbuf, wdn_bf, counts, slot_token, slot_w, out);
}
